// C2QAttention_22789096473044
// MI455X (gfx1250) — matrix operand order repaired and checked by reading, not compiled, not run on hardware
//
#include <hip/hip_runtime.h>
#include <math.h>

typedef __attribute__((ext_vector_type(16))) _Float16 v16h;
typedef __attribute__((ext_vector_type(8)))  float    v8f;
typedef __attribute__((ext_vector_type(4)))  float    f4;
typedef __attribute__((ext_vector_type(4)))  int      v4i;

#define NB 16
#define NC 4096
#define NQ 512
#define ND 128
#define QCHUNKS (NQ / 32)   // 16 K-chunks of 32
#define DTILES  (ND / 16)   // 8 N-tiles of 16
#define MTILES  2           // 32 C-rows per wave
#define MROWS   (MTILES * 16)
#define F16_ROW_BYTES 1040  // 512*2 + 16B pad -> 4-bank rotation per row
#define BCHUNK_BYTES (DTILES * 1024)   // 8KB of B fragments per K-chunk

#if __has_builtin(__builtin_amdgcn_global_load_async_to_lds_b128)
#define HAVE_ASYNC_LDS 1
#else
#define HAVE_ASYNC_LDS 0
#endif

#if __has_builtin(__builtin_amdgcn_s_wait_asynccnt)
#define WAIT_ASYNC(n) __builtin_amdgcn_s_wait_asynccnt(n)
#else
#define WAIT_ASYNC(n) asm volatile("s_wait_asynccnt %0" :: "i"(n) : "memory")
#endif

// exact parameter types per hipcc diagnostics:
//   arg0: AS(1) int4*   (global source)
//   arg1: AS(3) int4*   (LDS destination)
typedef __attribute__((address_space(1))) v4i g_v4i;
typedef __attribute__((address_space(3))) v4i l_v4i;

// ---------------------------------------------------------------------------
// Pre-pass: encoded_question (f32 [B,Q,D]) -> f16 WMMA B-fragments in d_ws.
// Fragment (b, qc, t) = B tile K=qc*32..+31, N=t*16..+15, per-lane contiguous-k order (this kernel's choice) for
// v_wmma_f32_16x16x32_f16 (lane L: N=L%16, half=L/16; VGPR v: K=16*half+2v,+1).
// Stored as two 512B halves: [half0: lane*16][half1: 512 + lane*16] so both
// async-copy source and LDS-staged ds_load reads are contiguous/conflict-free.
// ---------------------------------------------------------------------------
__global__ __launch_bounds__(32)
void c2q_prep_b(const float* __restrict__ eq, uint4* __restrict__ ws) {
    const int frag = blockIdx.x;            // 0 .. NB*QCHUNKS*DTILES-1
    const int b    = frag >> 7;
    const int rem  = frag & 127;
    const int qc   = rem >> 3;
    const int t    = rem & 7;
    const int lane = threadIdx.x;
    const int n    = lane & 15;
    const int h    = lane >> 4;
    const int q0   = qc * 32 + h * 16;
    const int d    = t * 16 + n;

    const float* src = eq + ((size_t)b * NQ + q0) * ND + d;
    union { v16h h16; uint4 u[2]; } f;
#pragma unroll
    for (int j = 0; j < 16; ++j)
        f.h16[j] = (_Float16)src[(size_t)j * ND];

    uint4* dst = ws + (size_t)frag * 64;    // 1024 B per fragment
    dst[lane]      = f.u[0];                // half0 block
    dst[32 + lane] = f.u[1];                // half1 block
}

// ---------------------------------------------------------------------------
__device__ __forceinline__
void softmax_row_store(f4 v[4], unsigned char* smrow, int lane) {
    float mx = -INFINITY;
#pragma unroll
    for (int i = 0; i < 4; ++i)
        mx = fmaxf(mx, fmaxf(fmaxf(v[i].x, v[i].y), fmaxf(v[i].z, v[i].w)));
#pragma unroll
    for (int off = 16; off >= 1; off >>= 1)
        mx = fmaxf(mx, __shfl_xor(mx, off, 32));

    float s = 0.0f;
#pragma unroll
    for (int i = 0; i < 4; ++i) {
        v[i].x = __expf(v[i].x - mx);
        v[i].y = __expf(v[i].y - mx);
        v[i].z = __expf(v[i].z - mx);
        v[i].w = __expf(v[i].w - mx);
        s += (v[i].x + v[i].y) + (v[i].z + v[i].w);
    }
#pragma unroll
    for (int off = 16; off >= 1; off >>= 1)
        s += __shfl_xor(s, off, 32);
    const float inv = 1.0f / s;   // fold softmax normalization into A matrix

#pragma unroll
    for (int i = 0; i < 4; ++i) {
        union { _Float16 h[4]; uint2 u; } pk;
        pk.h[0] = (_Float16)(v[i].x * inv);
        pk.h[1] = (_Float16)(v[i].y * inv);
        pk.h[2] = (_Float16)(v[i].z * inv);
        pk.h[3] = (_Float16)(v[i].w * inv);
        *(uint2*)(smrow + i * 256 + lane * 8) = pk.u;
    }
}

// ---------------------------------------------------------------------------
// Main kernel: one wave32 per 32-row C-tile.
// Pass 1: fused streaming softmax -> f16 probs in LDS (A-fragment layout).
// Pass 2: 32x512 @ 512x128 with v_wmma_f32_16x16x32_f16; B fragments staged
//         global->LDS with async copies, double-buffered one chunk ahead;
//         B tiles additionally pipelined one deep LDS->VGPR.
// ---------------------------------------------------------------------------
__global__ __launch_bounds__(32)
void c2q_attn(const float* __restrict__ sim, const uint4* __restrict__ bws,
              float* __restrict__ out) {
    __shared__ __align__(16) unsigned char smA[MROWS * F16_ROW_BYTES];
#if HAVE_ASYNC_LDS
    __shared__ __align__(16) unsigned char smB[2 * BCHUNK_BYTES];
#endif

    const int lane = threadIdx.x;
    const int blk  = blockIdx.x;            // NB * (NC/MROWS) blocks
    const int b    = blk >> 7;
    const int ct   = blk & 127;
    const size_t rowbase = ((size_t)b * NC + (size_t)ct * MROWS) * NQ;

    // ---- Pass 1: streaming softmax, 4 rows (16 b128, 8KB) in flight.
    //      NT hint: sim is read exactly once; don't thrash B-fragments in L2.
    for (int r = 0; r < MROWS; r += 4) {
        f4 v[4][4];
#pragma unroll
        for (int rr = 0; rr < 4; ++rr) {
            const f4* rp = (const f4*)(sim + rowbase + (size_t)(r + rr) * NQ);
#pragma unroll
            for (int i = 0; i < 4; ++i)
                v[rr][i] = __builtin_nontemporal_load(rp + i * 32 + lane);
        }
#pragma unroll
        for (int rr = 0; rr < 4; ++rr)
            softmax_row_store(v[rr], smA + (size_t)(r + rr) * F16_ROW_BYTES, lane);
    }
    // single wave: DS ops are in program order; no barrier needed.

    // ---- Pass 2
    const int m = lane & 15;
    const int h = lane >> 4;

    v8f acc[MTILES][DTILES];
#pragma unroll
    for (int mt = 0; mt < MTILES; ++mt)
#pragma unroll
        for (int t = 0; t < DTILES; ++t) {
            v8f z = {0.f, 0.f, 0.f, 0.f, 0.f, 0.f, 0.f, 0.f};
            acc[mt][t] = z;
        }

    const uint4* bbase = bws + (size_t)b * QCHUNKS * DTILES * 64;

#if HAVE_ASYNC_LDS
    // prologue: async-stage chunk 0 into buffer 0 (16 instrs, ASYNCcnt += 16)
    {
        const uint4* bq = bbase;
#pragma unroll
        for (int t = 0; t < DTILES; ++t) {
            __builtin_amdgcn_global_load_async_to_lds_b128(
                (g_v4i*)(bq + t * 64 + lane),
                (l_v4i*)(smB + t * 1024 + lane * 16), 0, 0);
            __builtin_amdgcn_global_load_async_to_lds_b128(
                (g_v4i*)(bq + t * 64 + 32 + lane),
                (l_v4i*)(smB + t * 1024 + 512 + lane * 16), 0, 0);
        }
    }
#endif

    // unroll=2 only: covers both double-buffer parities so all async LDS
    // destination addresses stay loop-invariant; full unroll was observed to
    // materialize 256 unique address VGPRs (>256 regs, s_set_vgpr_msb churn).
#pragma unroll 2
    for (int qc = 0; qc < QCHUNKS; ++qc) {
        // A fragments: lane L row M=L%16; K range [16h,16h+16)
        union { v16h h16; uint4 u[2]; } af[MTILES];
#pragma unroll
        for (int mt = 0; mt < MTILES; ++mt) {
            const unsigned char* arow =
                smA + (size_t)(mt * 16 + m) * F16_ROW_BYTES + qc * 64 + h * 32;
            af[mt].u[0] = *(const uint4*)(arow);
            af[mt].u[1] = *(const uint4*)(arow + 16);
        }

#if HAVE_ASYNC_LDS
        // issue next chunk's async copies into the other buffer, then wait
        // for the current chunk's 16 copies (allow 16 outstanding).
        if (qc < QCHUNKS - 1) {
            const uint4* bq = bbase + (size_t)(qc + 1) * DTILES * 64;
            unsigned char* dstb = smB + ((qc + 1) & 1) * BCHUNK_BYTES;
#pragma unroll
            for (int t = 0; t < DTILES; ++t) {
                __builtin_amdgcn_global_load_async_to_lds_b128(
                    (g_v4i*)(bq + t * 64 + lane),
                    (l_v4i*)(dstb + t * 1024 + lane * 16), 0, 0);
                __builtin_amdgcn_global_load_async_to_lds_b128(
                    (g_v4i*)(bq + t * 64 + 32 + lane),
                    (l_v4i*)(dstb + t * 1024 + 512 + lane * 16), 0, 0);
            }
            WAIT_ASYNC(16);
        } else {
            WAIT_ASYNC(0);
        }

        // B tiles: one-deep LDS->VGPR pipeline so each WMMA pair issues with
        // the next tile's ds_loads already in flight (s_wait_dscnt 2, not 0).
        const unsigned char* bbuf = smB + (qc & 1) * BCHUNK_BYTES;
        union { v16h h16; uint4 u[2]; } bcur, bnxt;
        bcur.u[0] = *(const uint4*)(bbuf + lane * 16);
        bcur.u[1] = *(const uint4*)(bbuf + 512 + lane * 16);
#pragma unroll
        for (int t = 0; t < DTILES; ++t) {
            if (t < DTILES - 1) {
                bnxt.u[0] = *(const uint4*)(bbuf + (t + 1) * 1024 + lane * 16);
                bnxt.u[1] = *(const uint4*)(bbuf + (t + 1) * 1024 + 512 + lane * 16);
            }
            acc[0][t] = __builtin_amdgcn_wmma_f32_16x16x32_f16(
                false, af[0].h16, false, bcur.h16, (short)0, acc[0][t], false, false);
            acc[1][t] = __builtin_amdgcn_wmma_f32_16x16x32_f16(
                false, af[1].h16, false, bcur.h16, (short)0, acc[1][t], false, false);
            if (t < DTILES - 1) bcur = bnxt;
        }
#else
        // fallback: synchronous B-fragment loads straight to VGPRs
        const uint4* bq = bbase + (size_t)qc * DTILES * 64;
#pragma unroll
        for (int t = 0; t < DTILES; ++t) {
            union { v16h h16; uint4 u[2]; } bf;
            bf.u[0] = bq[t * 64 + lane];
            bf.u[1] = bq[t * 64 + 32 + lane];
            acc[0][t] = __builtin_amdgcn_wmma_f32_16x16x32_f16(
                false, af[0].h16, false, bf.h16, (short)0, acc[0][t], false, false);
            acc[1][t] = __builtin_amdgcn_wmma_f32_16x16x32_f16(
                false, af[1].h16, false, bf.h16, (short)0, acc[1][t], false, false);
        }
#endif
    }

    // ---- Store (NT: written once, never re-read):
    //      lane L -> col N=L%16; VGPR r -> row M = mt*16 + r + 8*(L/16)
    float* op = out + ((size_t)b * NC + (size_t)ct * MROWS) * ND;
#pragma unroll
    for (int mt = 0; mt < MTILES; ++mt)
#pragma unroll
        for (int t = 0; t < DTILES; ++t) {
            const int d = t * 16 + m;
#pragma unroll
            for (int r = 0; r < 8; ++r) {
                __builtin_nontemporal_store(
                    acc[mt][t][r],
                    op + (size_t)(mt * 16 + r + 8 * h) * ND + d);
            }
        }
}

// ---------------------------------------------------------------------------
extern "C" void kernel_launch(void* const* d_in, const int* in_sizes, int n_in,
                              void* d_out, int out_size, void* d_ws, size_t ws_size,
                              hipStream_t stream) {
    const float* sim = (const float*)d_in[0];   // [16, 4096, 512] f32
    const float* eq  = (const float*)d_in[1];   // [16, 512, 128] f32
    float*       out = (float*)d_out;           // [16, 4096, 128] f32

    // B-fragment scratch: NB*QCHUNKS*DTILES fragments * 1024 B = 2 MB
    c2q_prep_b<<<NB * QCHUNKS * DTILES, 32, 0, stream>>>(eq, (uint4*)d_ws);
    c2q_attn<<<NB * (NC / MROWS), 32, 0, stream>>>(sim, (const uint4*)d_ws, out);
}
